// A2aSparseMLP_72310069396104
// MI455X (gfx1250) — compile-verified, matrix operand order checked by reading, not run on hardware
//
#include <hip/hip_runtime.h>
#include <hip/hip_bf16.h>
#include <math.h>

#define ALPHA 1.702f
#define LIMIT 7.0f

typedef __bf16 v16bf __attribute__((ext_vector_type(16)));
typedef __bf16 v8bf  __attribute__((ext_vector_type(8)));
typedef __bf16 v4bf  __attribute__((ext_vector_type(4)));
typedef float  v8f   __attribute__((ext_vector_type(8)));

static constexpr int S = 2048;
static constexpr int H = 1024;
static constexpr int I = 1024;
static constexpr int E = 8;
static constexpr int F = 2 * I;        // 2048 gate_up columns (interleaved gate/up)
static constexpr int OUT_OFF = S * H;  // router_scores offset in d_out
static constexpr int MAXTOK = S;       // per-expert token list capacity
static constexpr int MT = 32;          // tokens per workgroup (2 WMMA row subtiles)

union ABReg { v16bf v; v8bf h[2]; };

// ---------------- zero output + counters ----------------
__global__ void zero_init(float* __restrict__ out, int* __restrict__ cnt) {
  size_t idx = (size_t)blockIdx.x * blockDim.x + threadIdx.x;
  size_t n = (size_t)S * H;
  for (size_t i = idx; i < n; i += (size_t)gridDim.x * blockDim.x) out[i] = 0.0f;
  if (blockIdx.x == 0 && threadIdx.x < E) cnt[threadIdx.x] = 0;
}

// ---------------- router: one wave per token ----------------
__global__ __launch_bounds__(256)
void router_kernel(const float* __restrict__ hs,
                   const float* __restrict__ rw,
                   const float* __restrict__ rb,
                   float* __restrict__ scores,
                   int* __restrict__ cnt,
                   int* __restrict__ tokList,
                   float* __restrict__ wtList) {
  int wid  = threadIdx.x >> 5;
  int lane = threadIdx.x & 31;
  int t = blockIdx.x * (blockDim.x >> 5) + wid;
  if (t >= S) return;

  float acc[E];
#pragma unroll
  for (int e = 0; e < E; ++e) acc[e] = 0.0f;
  for (int h = lane; h < H; h += 32) {
    float x = hs[(size_t)t * H + h];
#pragma unroll
    for (int e = 0; e < E; ++e) acc[e] += x * rw[(size_t)h * E + e];
  }
#pragma unroll
  for (int off = 16; off > 0; off >>= 1) {
#pragma unroll
    for (int e = 0; e < E; ++e) acc[e] += __shfl_xor(acc[e], off, 32);
  }

  if (lane == 0) {
#pragma unroll
    for (int e = 0; e < E; ++e) acc[e] += rb[e];
    int i0 = 0; float v0 = acc[0];
#pragma unroll
    for (int e = 1; e < E; ++e) { if (acc[e] > v0) { v0 = acc[e]; i0 = e; } }
    int i1 = (i0 == 0) ? 1 : 0; float v1 = acc[i1];
#pragma unroll
    for (int e = 0; e < E; ++e) {
      if (e != i0 && acc[e] > v1) { v1 = acc[e]; i1 = e; }
    }
    float e1 = __expf(v1 - v0);
    float w0 = 1.0f / (1.0f + e1);
    float w1 = e1 * w0;
#pragma unroll
    for (int e = 0; e < E; ++e) {
      float s = (e == i0) ? w0 : ((e == i1) ? w1 : 0.0f);
      scores[(size_t)t * E + e] = s;
    }
    int p0 = atomicAdd(&cnt[i0], 1);
    tokList[i0 * MAXTOK + p0] = t;
    wtList[i0 * MAXTOK + p0] = w0;
    int p1 = atomicAdd(&cnt[i1], 1);
    tokList[i1 * MAXTOK + p1] = t;
    wtList[i1 * MAXTOK + p1] = w1;
  }
}

// ---------------- fused expert MLP: 32 tokens x 1 expert per block ----------------
__global__ __launch_bounds__(256)
void moe_kernel(const float* __restrict__ hs,
                const float* __restrict__ gup,
                const float* __restrict__ gub,
                const float* __restrict__ dp,
                const float* __restrict__ db,
                const int* __restrict__ cnt,
                const int* __restrict__ tokList,
                const float* __restrict__ wtList,
                float* __restrict__ out) {
  const int e    = blockIdx.y;
  const int tile = blockIdx.x;
  const int nTok = cnt[e];
  if (tile * MT >= nTok) return;
  int nv = nTok - tile * MT; if (nv > MT) nv = MT;

  __shared__ __align__(16) __bf16 ldsA[MT * H];    // 64 KB: gathered hidden states
  __shared__ __align__(16) __bf16 ldsAct[MT * I];  // 64 KB: activated intermediates
  __shared__ int   sTok[MT];
  __shared__ float sWt[MT];

  if (threadIdx.x < MT) {
    int m = threadIdx.x;
    sTok[m] = (m < nv) ? tokList[e * MAXTOK + tile * MT + m] : 0;
    sWt[m]  = (m < nv) ? wtList[e * MAXTOK + tile * MT + m] : 0.0f;
  }
  __syncthreads();

  // ---- phase 0: gather + fp32->bf16 (float4 loads, b64 LDS stores) ----
  for (int idx = threadIdx.x; idx < MT * (H / 4); idx += blockDim.x) {
    int row = idx >> 8;                 // H/4 == 256
    int c4  = (idx & 255) * 4;
    float4 v = make_float4(0.f, 0.f, 0.f, 0.f);
    if (row < nv) v = *(const float4*)(hs + (size_t)sTok[row] * H + c4);
    v4bf o;
    o[0] = (__bf16)v.x; o[1] = (__bf16)v.y; o[2] = (__bf16)v.z; o[3] = (__bf16)v.w;
    *(v4bf*)(ldsA + row * H + c4) = o;
  }
  __syncthreads();

  const int wid   = threadIdx.x >> 5;
  const int lane  = threadIdx.x & 31;
  const int r15   = lane & 15;              // A row / B-C column index within tile
  const int moff  = (lane & 16) ? 8 : 0;    // C row offset for hi half-wave
  const int khalf = (lane & 16) ? 8 : 0;    // K offset for hi half-wave

  // ---- phase 1: gate_up GEMM + activation ----
  // Per wave-tile: 16 activated columns (32 interleaved gate_up columns) x 32 rows.
  // One float2 weight load feeds the gate and up WMMA B operands, and each B tile
  // is reused across 2 A row-subtiles. Activation is lane-local (no shuffles).
  const float* Wgu = gup + (size_t)e * H * F;
  for (int nt = wid; nt < I / 16; nt += 8) {
    const int a0 = nt * 16;
    const int cg = 2 * (a0 + r15);          // this lane's gate column in memory
    v8f accg0 = {}, accu0 = {}, accg1 = {}, accu1 = {};
    for (int kb = 0; kb < H / 32; ++kb) {
      const int k0 = kb * 32 + khalf;
      ABReg aA, aB, bg, bu;
      aA.h[0] = *(const v8bf*)(ldsA + r15 * H + k0);
      aA.h[1] = *(const v8bf*)(ldsA + r15 * H + k0 + 16);
      aB.h[0] = *(const v8bf*)(ldsA + (r15 + 16) * H + k0);
      aB.h[1] = *(const v8bf*)(ldsA + (r15 + 16) * H + k0 + 16);
#pragma unroll
      for (int j = 0; j < 8; ++j) {
        float2 w0 = *(const float2*)(Wgu + (size_t)(k0 + j) * F + cg);
        float2 w1 = *(const float2*)(Wgu + (size_t)(k0 + 16 + j) * F + cg);
        bg.v[j]     = (__bf16)w0.x;
        bu.v[j]     = (__bf16)w0.y;
        bg.v[j + 8] = (__bf16)w1.x;
        bu.v[j + 8] = (__bf16)w1.y;
      }
      accg0 = __builtin_amdgcn_wmma_f32_16x16x32_bf16(false, aA.v, false, bg.v,
                                                      (short)0, accg0, false, false);
      accu0 = __builtin_amdgcn_wmma_f32_16x16x32_bf16(false, aA.v, false, bu.v,
                                                      (short)0, accu0, false, false);
      accg1 = __builtin_amdgcn_wmma_f32_16x16x32_bf16(false, aB.v, false, bg.v,
                                                      (short)0, accg1, false, false);
      accu1 = __builtin_amdgcn_wmma_f32_16x16x32_bf16(false, aB.v, false, bu.v,
                                                      (short)0, accu1, false, false);
    }
    float2 bias = *(const float2*)(gub + (size_t)e * F + cg);
#pragma unroll
    for (int sub = 0; sub < 2; ++sub) {
      const v8f& ag = sub ? accg1 : accg0;
      const v8f& au = sub ? accu1 : accu0;
#pragma unroll
      for (int r = 0; r < 8; ++r) {
        float g = ag[r] + bias.x;
        float u = au[r] + bias.y;
        g = fminf(g, LIMIT);
        u = fminf(fmaxf(u, -LIMIT), LIMIT);
        float glu = g * __builtin_amdgcn_rcpf(1.0f + __expf(-ALPHA * g));
        float act = (u + 1.0f) * glu;
        ldsAct[(r + moff + 16 * sub) * I + a0 + r15] = (__bf16)act;
      }
    }
  }
  __syncthreads();

  // ---- phase 2: down GEMM + bias + weighted scatter ----
  // 32 output columns per wave tile (even cols acc*, odd cols acc*1); each float2
  // down_proj load feeds 4 WMMAs (2 col parities x 2 row subtiles).
  const float* Wd = dp + (size_t)e * I * H;
  for (int nt = wid; nt < H / 32; nt += 8) {
    const int h0 = nt * 32;
    const int he = h0 + 2 * r15;            // this lane's even output column
    v8f acc00 = {}, acc01 = {}, acc10 = {}, acc11 = {};
    for (int kb = 0; kb < I / 32; ++kb) {
      const int k0 = kb * 32 + khalf;
      ABReg aA, aB, b0, b1;
      aA.h[0] = *(const v8bf*)(ldsAct + r15 * I + k0);
      aA.h[1] = *(const v8bf*)(ldsAct + r15 * I + k0 + 16);
      aB.h[0] = *(const v8bf*)(ldsAct + (r15 + 16) * I + k0);
      aB.h[1] = *(const v8bf*)(ldsAct + (r15 + 16) * I + k0 + 16);
#pragma unroll
      for (int j = 0; j < 8; ++j) {
        float2 w0 = *(const float2*)(Wd + (size_t)(k0 + j) * H + he);
        float2 w1 = *(const float2*)(Wd + (size_t)(k0 + 16 + j) * H + he);
        b0.v[j]     = (__bf16)w0.x;
        b1.v[j]     = (__bf16)w0.y;
        b0.v[j + 8] = (__bf16)w1.x;
        b1.v[j + 8] = (__bf16)w1.y;
      }
      acc00 = __builtin_amdgcn_wmma_f32_16x16x32_bf16(false, aA.v, false, b0.v,
                                                      (short)0, acc00, false, false);
      acc01 = __builtin_amdgcn_wmma_f32_16x16x32_bf16(false, aA.v, false, b1.v,
                                                      (short)0, acc01, false, false);
      acc10 = __builtin_amdgcn_wmma_f32_16x16x32_bf16(false, aB.v, false, b0.v,
                                                      (short)0, acc10, false, false);
      acc11 = __builtin_amdgcn_wmma_f32_16x16x32_bf16(false, aB.v, false, b1.v,
                                                      (short)0, acc11, false, false);
    }
    float2 bias = *(const float2*)(db + (size_t)e * H + he);
#pragma unroll
    for (int sub = 0; sub < 2; ++sub) {
      const v8f& a0r = sub ? acc10 : acc00;
      const v8f& a1r = sub ? acc11 : acc01;
#pragma unroll
      for (int r = 0; r < 8; ++r) {
        int m = r + moff + 16 * sub;
        if (m < nv) {
          float w = sWt[m];
          float* dst = out + (size_t)sTok[m] * H + he;
          atomicAdd(dst,     (a0r[r] + bias.x) * w);
          atomicAdd(dst + 1, (a1r[r] + bias.y) * w);
        }
      }
    }
  }
}

extern "C" void kernel_launch(void* const* d_in, const int* in_sizes, int n_in,
                              void* d_out, int out_size, void* d_ws, size_t ws_size,
                              hipStream_t stream) {
  (void)in_sizes; (void)n_in; (void)out_size; (void)ws_size;
  const float* hs  = (const float*)d_in[0];
  const float* rw  = (const float*)d_in[1];
  const float* rb  = (const float*)d_in[2];
  const float* gup = (const float*)d_in[3];
  const float* gub = (const float*)d_in[4];
  const float* dp  = (const float*)d_in[5];
  const float* db  = (const float*)d_in[6];

  float* out    = (float*)d_out;
  float* scores = out + OUT_OFF;

  char* ws = (char*)d_ws;
  int*   cnt     = (int*)ws;                                   // 8 ints
  int*   tokList = (int*)(ws + 64);                            // E*S ints
  float* wtList  = (float*)(ws + 64 + sizeof(int) * E * MAXTOK);

  zero_init<<<512, 256, 0, stream>>>(out, cnt);
  router_kernel<<<S / 8, 256, 0, stream>>>(hs, rw, rb, scores, cnt, tokList, wtList);
  dim3 grid(S / MT, E);
  moe_kernel<<<grid, 256, 0, stream>>>(hs, gup, gub, dp, db, cnt, tokList, wtList, out);
}
